// Conv2d_87737591922985
// MI455X (gfx1250) — matrix operand order repaired and checked by reading, not compiled, not run on hardware
//
#include <hip/hip_runtime.h>

// ---------------------------------------------------------------------------
// Implicit-GEMM 3x3 conv (pad=1, stride=1) for MI455X / gfx1250.
//   M = B*H*W = 100352, N = COUT = 256, K = CIN*9 = 2304
//   Prep kernels: NCHW f32 -> NHWC bf16 (A), weights -> bf16 [ks][n][ci] (B).
//   Main kernel: pure-copy LDS staging (async-to-LDS when available),
//   bf16 WMMA 16x16x32 with f32 accumulate, bias folded into acc init,
//   vectorized b128 epilogue stores.
// ---------------------------------------------------------------------------

typedef __attribute__((ext_vector_type(16))) __bf16 v16bf;
typedef __attribute__((ext_vector_type(8)))  __bf16 v8bf;
typedef __attribute__((ext_vector_type(8)))  float  v8f;
typedef __attribute__((ext_vector_type(4)))  int    v4i;

#define B_    32
#define CIN_  256
#define COUT_ 256
#define H_    56
#define W_    56
#define HW_   (H_ * W_)
#define M_TOTAL (B_ * HW_)            // 100352

#define MT  128                       // M tile per workgroup
#define NT  256                       // N tile per workgroup (all of COUT)
#define KSTEPS (9 * (CIN_ / 32))      // 72 k-steps of 32
#define LDA 40                        // padded LDS row stride (bf16), 80 B
#define LDB 40

#define WSA_ELEMS ((size_t)B_ * H_ * W_ * CIN_)          // NHWC bf16 input
#define WSB_ELEMS ((size_t)KSTEPS * NT * 32)             // repacked bf16 weights
#define WSA_BYTES (WSA_ELEMS * 2)                        // 51,380,224

#if defined(__AMDGCN__) && __has_builtin(__builtin_amdgcn_global_load_async_to_lds_b128)
#define HAVE_ASYNC 1
#else
#define HAVE_ASYNC 0
#endif

#if HAVE_ASYNC
typedef v4i __attribute__((address_space(1))) gv4i;   // global v4i
typedef v4i __attribute__((address_space(3))) lv4i;   // LDS v4i
#endif

__device__ __forceinline__ unsigned short f2bf(float f) {
    unsigned u = __float_as_uint(f);
    unsigned r = u + 0x7FFFu + ((u >> 16) & 1u);   // round-to-nearest-even
    return (unsigned short)(r >> 16);
}

// 16-byte global -> LDS copy (async DMA when the toolchain exposes it)
__device__ __forceinline__ void cp16_g2l(const unsigned short* g, unsigned short* l) {
#if HAVE_ASYNC
    __builtin_amdgcn_global_load_async_to_lds_b128((gv4i*)g, (lv4i*)l, 0, 0);
#else
    *(uint4*)l = *(const uint4*)g;
#endif
}

__device__ __forceinline__ void wait_async0() {
#if HAVE_ASYNC
#if __has_builtin(__builtin_amdgcn_s_wait_asynccnt)
    __builtin_amdgcn_s_wait_asynccnt(0);
#else
    asm volatile("s_wait_asynccnt 0x0" ::: "memory");
#endif
#endif
}

// ---- prep 1: NCHW f32 -> NHWC bf16 -----------------------------------------
__global__ __launch_bounds__(256)
void nchw_to_nhwc_bf16(const float* __restrict__ in, unsigned short* __restrict__ outA)
{
    const size_t idx = (size_t)blockIdx.x * 256 + threadIdx.x;   // NHWC index
    const int c = (int)(idx & (CIN_ - 1));
    const size_t p = idx >> 8;                                   // b*HW + ohw
    const int b = (int)(p / HW_);
    const int ohw = (int)(p - (size_t)b * HW_);
    outA[idx] = f2bf(in[((size_t)b * CIN_ + c) * HW_ + ohw]);
}

// ---- prep 2: weights OIHW f32 -> bf16 [ks][n][ci32] ------------------------
__global__ __launch_bounds__(256)
void repack_weights_bf16(const float* __restrict__ w, unsigned short* __restrict__ outB)
{
    const size_t idx = (size_t)blockIdx.x * 256 + threadIdx.x;
    const int c  = (int)(idx & 31);
    const int n  = (int)((idx >> 5) & (NT - 1));
    const int ks = (int)(idx >> 13);           // 0..71
    const int tap = ks >> 3;                   // kh*3+kw
    const int cb  = ks & 7;
    outB[idx] = f2bf(w[((size_t)n * CIN_ + (cb << 5) + c) * 9 + tap]);
}

// ---- main kernel ------------------------------------------------------------
__global__ __launch_bounds__(256)
void conv3x3_wmma_bf16(const unsigned short* __restrict__ nhwc,
                       const unsigned short* __restrict__ wks,
                       const float* __restrict__ bias,
                       float* __restrict__ out)
{
    __shared__ alignas(16) unsigned short ldsA[2][MT * LDA];  // [m][ci] bf16
    __shared__ alignas(16) unsigned short ldsB[2][NT * LDB];  // [n][ci] bf16

    const int tid   = threadIdx.x;
    const int lane  = tid & 31;
    const int wave  = tid >> 5;
    const int waveM = wave & 3;        // 4 waves along M (32 rows each)
    const int waveN = wave >> 2;       // 2 waves along N (128 cols each)
    const int m0    = blockIdx.x * MT;

    // A-loader coordinates (fixed per thread): row m_loc, 16-ci chunk
    const int amloc = tid & (MT - 1);          // 0..127
    const int chunk = tid >> 7;                // 0/1 -> ci sub-chunk of 16
    const int am  = m0 + amloc;
    const int ab  = am / HW_;
    const int ahw = am - ab * HW_;
    const int ah  = ahw / W_;
    const int aw  = ahw - ah * W_;
    const unsigned short* aRowBase =
        nhwc + ((size_t)ab * H_ * W_) * CIN_;  // + (ih*W+iw)*CIN + ci later

    const int row16  = lane & 15;
    const int coBase = waveN * 128 + row16;

    // accumulators initialized with bias (bias broadcast over M)
    v8f acc[2][8];
#pragma unroll
    for (int ni = 0; ni < 8; ++ni) {
        const float bv = bias[coBase + ni * 16];
        const v8f bb = (v8f){bv, bv, bv, bv, bv, bv, bv, bv};
#pragma unroll
        for (int mi = 0; mi < 2; ++mi) acc[mi][ni] = bb;
    }

    // ---- stage one 32-wide K slab (pure copy) into buffer bufI ----
    auto stage = [&](int ks, int bufI) {
        const int tap = ks >> 3;
        const int cb  = ks & 7;
        const int kh  = tap / 3;
        const int kw  = tap - kh * 3;

        // A: 128 rows x 32 ci; this thread: row amloc, ci [cb*32+chunk*16, +16)
        const int ih = ah + kh - 1;
        const int iw = aw + kw - 1;
        unsigned short* da = &ldsA[bufI][amloc * LDA + chunk * 16];
        if ((unsigned)ih < (unsigned)H_ && (unsigned)iw < (unsigned)W_) {
            const unsigned short* sa =
                aRowBase + ((size_t)(ih * W_ + iw)) * CIN_ + (cb << 5) + chunk * 16;
            cp16_g2l(sa,     da);
            cp16_g2l(sa + 8, da + 8);
        } else {
            const uint4 z = {0u, 0u, 0u, 0u};
            *(uint4*)da       = z;              // zero padding (halo)
            *(uint4*)(da + 8) = z;
        }

        // B: 256 n x 32 ci; this thread copies row n = tid (64 contiguous B)
        const unsigned short* sb = wks + ((size_t)ks * NT + tid) * 32;
        unsigned short* db = &ldsB[bufI][tid * LDB];
        cp16_g2l(sb,      db);
        cp16_g2l(sb + 8,  db + 8);
        cp16_g2l(sb + 16, db + 16);
        cp16_g2l(sb + 24, db + 24);
    };

    stage(0, 0);
    wait_async0();
    __syncthreads();

    const int kA = (lane < 16) ? 0 : 8;     // A frag K halves per ISA layout
    const int kB = (lane < 16) ? 0 : 8;     // B frag K halves per ISA layout

    for (int ks = 0; ks < KSTEPS; ++ks) {
        const int buf = ks & 1;
        if (ks + 1 < KSTEPS) stage(ks + 1, buf ^ 1);   // overlap with WMMA

        // A fragments: lane<16 holds K {0..7,16..23}, lane>=16 {8..15,24..31}
        v16bf afr[2];
#pragma unroll
        for (int mi = 0; mi < 2; ++mi) {
            const unsigned short* p =
                &ldsA[buf][(waveM * 32 + mi * 16 + row16) * LDA + kA];
            v8bf lo = *(const v8bf*)p;
            v8bf hi = *(const v8bf*)(p + 16);
            afr[mi] = __builtin_shufflevector(lo, hi,
                0, 1, 2, 3, 4, 5, 6, 7, 8, 9, 10, 11, 12, 13, 14, 15);
        }

        // B fragments: col = lane&15; lane<16 K {0..7,16..23}, lane>=16 {8..15,24..31}
        v16bf bfr[8];
#pragma unroll
        for (int ni = 0; ni < 8; ++ni) {
            const unsigned short* p =
                &ldsB[buf][(waveN * 128 + ni * 16 + row16) * LDB + kB];
            v8bf lo = *(const v8bf*)p;
            v8bf hi = *(const v8bf*)(p + 16);
            bfr[ni] = __builtin_shufflevector(lo, hi,
                0, 1, 2, 3, 4, 5, 6, 7, 8, 9, 10, 11, 12, 13, 14, 15);
        }

#pragma unroll
        for (int mi = 0; mi < 2; ++mi)
#pragma unroll
            for (int ni = 0; ni < 8; ++ni)
                acc[mi][ni] = __builtin_amdgcn_wmma_f32_16x16x32_bf16(
                    false, afr[mi], false, bfr[ni],
                    (short)0, acc[mi][ni], false, false);

        wait_async0();          // async LDS writes done before releasing barrier
        __syncthreads();
    }

    // ---- epilogue: per lane each v8f = 8 consecutive output floats ----------
    // D layout: VGPR j -> M = j (lanes 0-15) / j+8 (lanes 16-31), N = lane&15.
    // mm is a multiple of 8; batch boundary (3136) is a multiple of 8, so each
    // 8-float span stays inside one (b, co) plane -> contiguous, 32B aligned.
#pragma unroll
    for (int mi = 0; mi < 2; ++mi) {
        const int mm = m0 + waveM * 32 + mi * 16 + ((lane < 16) ? 0 : 8);
        const int ob  = mm / HW_;
        const int ohw = mm - ob * HW_;
        const size_t base = (size_t)ob * COUT_ * HW_ + ohw;
#pragma unroll
        for (int ni = 0; ni < 8; ++ni) {
            const int co = coBase + ni * 16;
            *(v8f*)(out + base + (size_t)co * HW_) = acc[mi][ni];
        }
    }
}

extern "C" void kernel_launch(void* const* d_in, const int* in_sizes, int n_in,
                              void* d_out, int out_size, void* d_ws, size_t ws_size,
                              hipStream_t stream) {
    (void)in_sizes; (void)n_in; (void)out_size; (void)ws_size;
    const float* inp  = (const float*)d_in[0];
    const float* wgt  = (const float*)d_in[1];
    const float* bias = (const float*)d_in[2];
    float* out = (float*)d_out;

    unsigned short* wsA = (unsigned short*)d_ws;                       // NHWC bf16
    unsigned short* wsB = (unsigned short*)((char*)d_ws + WSA_BYTES);  // weights

    nchw_to_nhwc_bf16<<<dim3((unsigned)(WSA_ELEMS / 256)), dim3(256), 0, stream>>>(inp, wsA);
    repack_weights_bf16<<<dim3((unsigned)(WSB_ELEMS / 256)), dim3(256), 0, stream>>>(wgt, wsB);
    conv3x3_wmma_bf16<<<dim3(M_TOTAL / MT), dim3(256), 0, stream>>>(wsA, wsB, bias, out);
}
